// BatchMoEProcessor_39616778338942
// MI455X (gfx1250) — matrix operand order repaired and checked by reading, not compiled, not run on hardware
//
#include <hip/hip_runtime.h>
#include <hip/hip_bf16.h>
#include <math.h>

#define BATCH 16384
#define DDIM 256
#define HDIM 512
#define TWOD 512

typedef __attribute__((ext_vector_type(16))) __bf16 v16bf;
typedef __attribute__((ext_vector_type(8)))  float v8f;

union Frag { uint4 u[2]; v16bf v; };

__device__ __forceinline__ unsigned short f2bf(float f) {
    unsigned u = __builtin_bit_cast(unsigned, f);
    u += 0x7FFFu + ((u >> 16) & 1u);   // round-to-nearest-even
    return (unsigned short)(u >> 16);
}

// ---------------------------------------------------------------------------
// Pack a row-major f32 weight [Kd, Nd] into bf16 WMMA B fragments in this file's contiguous k order (A loads match):
// per 32x16 (KxN) tile: lane = (k_in_tile/16)*16 + n_in_tile, elem = k_in_tile%16
// -> each lane's 16-bf16 fragment is 32 contiguous bytes, and a whole N-tile
// column (16 k-tiles) is a contiguous 16 KB slab.
// ---------------------------------------------------------------------------
__global__ void pack_w_kernel(const float* __restrict__ W,
                              unsigned short* __restrict__ Wpk,
                              int Kd, int Nd) {
    int t = blockIdx.x * blockDim.x + threadIdx.x;
    int total = Kd * Nd;
    if (t >= total) return;
    int kk = t / Nd, nn = t - kk * Nd;
    int kt = kk >> 5, kin = kk & 31;
    int nt = nn >> 4, nin = nn & 15;
    int lane = ((kin >> 4) << 4) | nin;
    int elem = kin & 15;
    size_t dst = ((size_t)(nt * (Kd >> 5) + kt) * 32 + lane) * 16 + elem;
    Wpk[dst] = f2bf(W[(size_t)kk * Nd + nn]);
}

// ---------------------------------------------------------------------------
// Gather: X[b, 0:256] = cur (bf16), X[b, 256:512] = masked neighbor mean.
// ---------------------------------------------------------------------------
__global__ void gather_kernel(const int* __restrict__ cells,
                              const float* __restrict__ full,
                              const int* __restrict__ idx,
                              const int* __restrict__ counts,
                              int K, unsigned short* __restrict__ X) {
    int b = blockIdx.x;
    int d = threadIdx.x;
    float cur = full[(size_t)cells[b] * DDIM + d];
    X[(size_t)b * TWOD + d] = f2bf(cur);
    int cnt = counts[b];             // uniform per block -> scalarized
    float acc = 0.f;
    for (int k = 0; k < cnt; ++k) {
        int nb = idx[(size_t)b * K + k];
        acc += full[(size_t)nb * DDIM + d];
    }
    float inv = 1.0f / (float)(cnt > 0 ? cnt : 1);
    X[(size_t)b * TWOD + DDIM + d] = f2bf(acc * inv);
}

// d_out starts as the residual: out = cur
__global__ void init_out_kernel(const int* __restrict__ cells,
                                const float* __restrict__ full,
                                float* __restrict__ out) {
    int b = blockIdx.x;
    int d = threadIdx.x;
    out[(size_t)b * DDIM + d] = full[(size_t)cells[b] * DDIM + d];
}

// ---------------------------------------------------------------------------
// Gates: softmax(cur @ Wg + bg) premultiplied by each expert's (count>0) mask.
// ---------------------------------------------------------------------------
__global__ void gates_kernel(const int* __restrict__ cells,
                             const float* __restrict__ full,
                             const float* __restrict__ Wg,
                             const float* __restrict__ bg,
                             const int* __restrict__ cnt0,
                             const int* __restrict__ cnt1,
                             const int* __restrict__ cnt2,
                             float* __restrict__ wgate) {
    int wave = threadIdx.x >> 5;
    int lane = threadIdx.x & 31;
    int b = blockIdx.x * 8 + wave;
    const float* cur = full + (size_t)cells[b] * DDIM;
    float p0 = 0.f, p1 = 0.f, p2 = 0.f;
#pragma unroll
    for (int i = 0; i < 8; ++i) {
        int d = lane * 8 + i;
        float c = cur[d];
        p0 += c * Wg[d * 3 + 0];
        p1 += c * Wg[d * 3 + 1];
        p2 += c * Wg[d * 3 + 2];
    }
#pragma unroll
    for (int m = 16; m; m >>= 1) {
        p0 += __shfl_xor(p0, m, 32);
        p1 += __shfl_xor(p1, m, 32);
        p2 += __shfl_xor(p2, m, 32);
    }
    if (lane == 0) {
        p0 += bg[0]; p1 += bg[1]; p2 += bg[2];
        float mx = fmaxf(p0, fmaxf(p1, p2));
        float e0 = __expf(p0 - mx), e1 = __expf(p1 - mx), e2 = __expf(p2 - mx);
        float inv = 1.f / (e0 + e1 + e2);
        wgate[b * 3 + 0] = (cnt0[b] > 0) ? e0 * inv : 0.f;
        wgate[b * 3 + 1] = (cnt1[b] > 0) ? e1 * inv : 0.f;
        wgate[b * 3 + 2] = (cnt2[b] > 0) ? e2 * inv : 0.f;
    }
}

// ---------------------------------------------------------------------------
// Async-copy a contiguous 64 KB B panel (4 N-tiles x K=512) global -> LDS
// via GLOBAL_LOAD_ASYNC_TO_LDS_B128 (ASYNCcnt-tracked); caller barriers.
// ---------------------------------------------------------------------------
__device__ __forceinline__ void async_copy_panel_64k(const unsigned short* gsrc,
                                                     unsigned short* lds,
                                                     int tid) {
    unsigned lbase = (unsigned)(unsigned long long)(uintptr_t)lds;
#pragma unroll
    for (int i = 0; i < 16; ++i) {
        unsigned off = (unsigned)(i * 4096 + tid * 16);
        unsigned long long g = (unsigned long long)(uintptr_t)gsrc + off;
        unsigned l = lbase + off;
        asm volatile("global_load_async_to_lds_b128 %0, %1, off"
                     :: "v"(l), "v"(g) : "memory");
    }
    asm volatile("s_wait_asynccnt 0x0" ::: "memory");
}

// ---------------------------------------------------------------------------
// Shared GEMM core: wave tile 32(M) x 64(N), K=512 (16 wmma k-steps).
// Each B fragment read from LDS feeds 2 WMMAs (A-register reuse) to stay
// under the WGP LDS bandwidth; A is software-pipelined one k-step ahead.
// ---------------------------------------------------------------------------
#define GEMM_CORE(arow0, arow1, Bs, lane, half8, acc0, acc1)                          \
    {                                                                                  \
        Frag a0c, a1c, a0n, a1n;                                                       \
        a0c.u[0] = *(const uint4*)((arow0) + 2 * (half8));                             \
        a0c.u[1] = *(const uint4*)((arow0) + 2 * (half8) + 8);                         \
        a1c.u[0] = *(const uint4*)((arow1) + 2 * (half8));                             \
        a1c.u[1] = *(const uint4*)((arow1) + 2 * (half8) + 8);                         \
        a0n = a0c; a1n = a1c;                                                          \
        for (int kt = 0; kt < 16; ++kt) {                                              \
            if (kt < 15) {                                                             \
                const unsigned short* ap0 = (arow0) + (kt + 1) * 32 + 2 * (half8);     \
                const unsigned short* ap1 = (arow1) + (kt + 1) * 32 + 2 * (half8);     \
                a0n.u[0] = *(const uint4*)(ap0);                                       \
                a0n.u[1] = *(const uint4*)(ap0 + 8);                                   \
                a1n.u[0] = *(const uint4*)(ap1);                                       \
                a1n.u[1] = *(const uint4*)(ap1 + 8);                                   \
            }                                                                          \
            const uint4* bsp = (const uint4*)(Bs) + kt * 64 + (lane) * 2;              \
            Frag b0, b1f, b2f, b3f;                                                    \
            b0.u[0]  = bsp[0];    b0.u[1]  = bsp[1];                                   \
            b1f.u[0] = bsp[1024]; b1f.u[1] = bsp[1025];                                \
            b2f.u[0] = bsp[2048]; b2f.u[1] = bsp[2049];                                \
            b3f.u[0] = bsp[3072]; b3f.u[1] = bsp[3073];                                \
            acc0[0] = __builtin_amdgcn_wmma_f32_16x16x32_bf16(false, a0c.v, false, b0.v,  (short)0, acc0[0], false, false); \
            acc1[0] = __builtin_amdgcn_wmma_f32_16x16x32_bf16(false, a1c.v, false, b0.v,  (short)0, acc1[0], false, false); \
            acc0[1] = __builtin_amdgcn_wmma_f32_16x16x32_bf16(false, a0c.v, false, b1f.v, (short)0, acc0[1], false, false); \
            acc1[1] = __builtin_amdgcn_wmma_f32_16x16x32_bf16(false, a1c.v, false, b1f.v, (short)0, acc1[1], false, false); \
            acc0[2] = __builtin_amdgcn_wmma_f32_16x16x32_bf16(false, a0c.v, false, b2f.v, (short)0, acc0[2], false, false); \
            acc1[2] = __builtin_amdgcn_wmma_f32_16x16x32_bf16(false, a1c.v, false, b2f.v, (short)0, acc1[2], false, false); \
            acc0[3] = __builtin_amdgcn_wmma_f32_16x16x32_bf16(false, a0c.v, false, b3f.v, (short)0, acc0[3], false, false); \
            acc1[3] = __builtin_amdgcn_wmma_f32_16x16x32_bf16(false, a1c.v, false, b3f.v, (short)0, acc1[3], false, false); \
            a0c = a0n; a1c = a1n;                                                      \
        }                                                                              \
    }

// ---------------------------------------------------------------------------
// GEMM1: H = tanh(X[16384,512] @ W1pk + b1), bf16 out.
// Block = 8 waves x (32x64) wave tiles = 256x64 block tile.
// ---------------------------------------------------------------------------
__global__ __launch_bounds__(256)
void gemm1_kernel(const unsigned short* __restrict__ X,
                  const unsigned short* __restrict__ Wpk,
                  const float* __restrict__ b1,
                  unsigned short* __restrict__ H) {
    __shared__ unsigned short Bs[32768];  // 64 KB panel
    int tid  = threadIdx.x;
    int wave = tid >> 5;
    int lane = tid & 31;
    int Mw = blockIdx.x * 256 + wave * 32;
    int half8 = (lane >> 4) * 8;

    async_copy_panel_64k(Wpk + (size_t)blockIdx.y * 32768, Bs, tid);
    __syncthreads();

    const unsigned short* arow0 = X + (size_t)(Mw + (lane & 15)) * TWOD;
    const unsigned short* arow1 = arow0 + (size_t)16 * TWOD;

    v8f acc0[4] = {};
    v8f acc1[4] = {};
    GEMM_CORE(arow0, arow1, Bs, lane, half8, acc0, acc1)

    int col0 = lane & 15;
#pragma unroll
    for (int t = 0; t < 4; ++t) {
        int col = blockIdx.y * 64 + t * 16 + col0;
        float bias = b1[col];
#pragma unroll
        for (int r = 0; r < 8; ++r) {
            int m0 = Mw + half8 + r;
            H[(size_t)m0 * HDIM + col]        = f2bf(tanhf(acc0[t][r] + bias));
            H[(size_t)(m0 + 16) * HDIM + col] = f2bf(tanhf(acc1[t][r] + bias));
        }
    }
}

// ---------------------------------------------------------------------------
// GEMM2: out += wgate[m,e] * (H[16384,512] @ W2pk + b2). N=256.
// ---------------------------------------------------------------------------
__global__ __launch_bounds__(256)
void gemm2_kernel(const unsigned short* __restrict__ Hb,
                  const unsigned short* __restrict__ Wpk,
                  const float* __restrict__ b2,
                  const float* __restrict__ wgate, int e,
                  float* __restrict__ out) {
    __shared__ unsigned short Bs[32768];  // 64 KB panel
    int tid  = threadIdx.x;
    int wave = tid >> 5;
    int lane = tid & 31;
    int Mw = blockIdx.x * 256 + wave * 32;
    int half8 = (lane >> 4) * 8;

    async_copy_panel_64k(Wpk + (size_t)blockIdx.y * 32768, Bs, tid);
    __syncthreads();

    const unsigned short* arow0 = Hb + (size_t)(Mw + (lane & 15)) * HDIM;
    const unsigned short* arow1 = arow0 + (size_t)16 * HDIM;

    v8f acc0[4] = {};
    v8f acc1[4] = {};
    GEMM_CORE(arow0, arow1, Bs, lane, half8, acc0, acc1)

    float wrow0[8], wrow1[8];
#pragma unroll
    for (int r = 0; r < 8; ++r) {
        wrow0[r] = wgate[(size_t)(Mw + half8 + r) * 3 + e];
        wrow1[r] = wgate[(size_t)(Mw + 16 + half8 + r) * 3 + e];
    }

    int col0 = lane & 15;
#pragma unroll
    for (int t = 0; t < 4; ++t) {
        int col = blockIdx.y * 64 + t * 16 + col0;
        float bias = b2[col];
#pragma unroll
        for (int r = 0; r < 8; ++r) {
            int m0 = Mw + half8 + r;
            out[(size_t)m0 * DDIM + col]        += wrow0[r] * (acc0[t][r] + bias);
            out[(size_t)(m0 + 16) * DDIM + col] += wrow1[r] * (acc1[t][r] + bias);
        }
    }
}

extern "C" void kernel_launch(void* const* d_in, const int* in_sizes, int n_in,
                              void* d_out, int out_size, void* d_ws, size_t ws_size,
                              hipStream_t stream) {
    const int*   cells = (const int*)d_in[0];
    const float* full  = (const float*)d_in[1];
    const int*   idx[3]  = {(const int*)d_in[2], (const int*)d_in[4], (const int*)d_in[6]};
    const int*   cnts[3] = {(const int*)d_in[3], (const int*)d_in[5], (const int*)d_in[7]};
    const float* W1[3] = {(const float*)d_in[8],  (const float*)d_in[12], (const float*)d_in[16]};
    const float* b1[3] = {(const float*)d_in[9],  (const float*)d_in[13], (const float*)d_in[17]};
    const float* W2[3] = {(const float*)d_in[10], (const float*)d_in[14], (const float*)d_in[18]};
    const float* b2[3] = {(const float*)d_in[11], (const float*)d_in[15], (const float*)d_in[19]};
    const float* Wg = (const float*)d_in[20];
    const float* bg = (const float*)d_in[21];
    float* out = (float*)d_out;
    const int Ks[3] = {26, 32, 16};

    // Workspace carve-out (~35 MB total)
    char* ws = (char*)d_ws;
    size_t off = 0;
    auto carve = [&](size_t bytes) -> char* {
        char* p = ws + off;
        off = (off + bytes + 255) & ~(size_t)255;
        return p;
    };
    unsigned short* W1pk[3];
    unsigned short* W2pk[3];
    for (int e = 0; e < 3; ++e) W1pk[e] = (unsigned short*)carve((size_t)512 * 512 * 2);
    for (int e = 0; e < 3; ++e) W2pk[e] = (unsigned short*)carve((size_t)512 * 256 * 2);
    unsigned short* X    = (unsigned short*)carve((size_t)BATCH * TWOD * 2);
    unsigned short* Hbuf = (unsigned short*)carve((size_t)BATCH * HDIM * 2);
    float*          wgate = (float*)carve((size_t)BATCH * 3 * 4);

    for (int e = 0; e < 3; ++e) {
        pack_w_kernel<<<(512 * 512 + 255) / 256, 256, 0, stream>>>(W1[e], W1pk[e], 512, 512);
        pack_w_kernel<<<(512 * 256 + 255) / 256, 256, 0, stream>>>(W2[e], W2pk[e], 512, 256);
    }

    init_out_kernel<<<BATCH, 256, 0, stream>>>(cells, full, out);
    gates_kernel<<<BATCH / 8, 256, 0, stream>>>(cells, full, Wg, bg,
                                                cnts[0], cnts[1], cnts[2], wgate);

    for (int e = 0; e < 3; ++e) {
        gather_kernel<<<BATCH, 256, 0, stream>>>(cells, full, idx[e], cnts[e], Ks[e], X);
        gemm1_kernel<<<dim3(BATCH / 256, HDIM / 64), 256, 0, stream>>>(X, W1pk[e], b1[e], Hbuf);
        gemm2_kernel<<<dim3(BATCH / 256, DDIM / 64), 256, 0, stream>>>(Hbuf, W2pk[e], b2[e],
                                                                       wgate, e, out);
    }
}
